// _UserModel_34359738368363
// MI455X (gfx1250) — matrix operand order repaired and checked by reading, not compiled, not run on hardware
//
#include <hip/hip_runtime.h>
#include <hip/hip_bf16.h>
#include <math.h>

typedef __attribute__((ext_vector_type(16))) _Float16 v16h;
typedef __attribute__((ext_vector_type(8)))  _Float16 v8h;
typedef __attribute__((ext_vector_type(8)))  float    v8f;

static constexpr int DIM   = 128;   // feature dim
static constexpr int NH    = 50;    // neighbors
static constexpr int BATCH = 16384;

__device__ __forceinline__ v8f wmma16x16x32(v16h a, v16h b, v8f c) {
  return __builtin_amdgcn_wmma_f32_16x16x32_f16(false, a, false, b, (short)0, c,
                                                false, false);
}

// A fragment: 16x32 f16 tile from LDS (row-major, stride DIM halves).
// ISA 16-bit A layout: lane<16 -> K {0..7,16..23}+k0 ; lane>=16 -> K {8..15,24..31}+k0.
__device__ __forceinline__ v16h load_fragA(const _Float16* __restrict__ base,
                                           int row0, int k0, int lane) {
  int m  = lane & 15;
  int kb = k0 + ((lane & 16) ? 8 : 0);
  const _Float16* p = base + (row0 + m) * DIM + kb;
  v8h lo = *(const v8h*)(p);        // K = kb .. kb+7
  v8h hi = *(const v8h*)(p + 16);   // K = kb+16 .. kb+23
  v16h a;
#pragma unroll
  for (int e = 0; e < 8; ++e) { a[e] = lo[e]; a[e + 8] = hi[e]; }
  return a;
}

// B fragment: 32x16 f16 weight tile. wt is the TRANSPOSED weight (N x K row-major
// f16): wt[n*DIM + k] == W[k][n]. Lane layout: lane<16 -> K {0..7,16..23}+k0 at col n,
// lane>=16 -> K {8..15,24..31}+k0 — two 16B global loads.
__device__ __forceinline__ v16h load_fragB(const _Float16* __restrict__ wt,
                                           int k0, int n0, int lane) {
  int n  = lane & 15;
  int kb = k0 + ((lane & 16) ? 8 : 0);
  const _Float16* p = wt + (size_t)(n0 + n) * DIM + kb;
  v8h lo = *(const v8h*)(p);
  v8h hi = *(const v8h*)(p + 16);
  v16h b;
#pragma unroll
  for (int e = 0; e < 8; ++e) { b[e] = lo[e]; b[e + 8] = hi[e]; }
  return b;
}

// Store 16x16 f32 accumulator to LDS f16 with bias + relu.
// C/D layout: lane<16 -> M r, lane>=16 -> M 8+r ; N = lane&15.
__device__ __forceinline__ void store_tile_relu(_Float16* __restrict__ dst,
                                                int row0, int n0,
                                                const float* __restrict__ bias,
                                                v8f c, int lane) {
  int n  = lane & 15;
  int m0 = (lane & 16) ? 8 : 0;
  float bv = bias[n0 + n];
#pragma unroll
  for (int r = 0; r < 8; ++r) {
    float v = c[r] + bv;
    dst[(row0 + m0 + r) * DIM + n0 + n] = (_Float16)(v > 0.f ? v : 0.f);
  }
}

// ---------------------------------------------------------------------------
// Kernel 0: fp32 weights -> transposed f16 weights in workspace.
// grid = 8*128 blocks, 128 threads. block -> (weight w, source row k).
// ---------------------------------------------------------------------------
__global__ __launch_bounds__(128) void prep_weights(
    const float* w0, const float* w1, const float* w2, const float* w3,
    const float* w4, const float* w5, const float* w6, const float* w7,
    _Float16* __restrict__ dst) {
  const float* srcs[8] = {w0, w1, w2, w3, w4, w5, w6, w7};
  int w = blockIdx.x >> 7;
  int k = blockIdx.x & 127;
  int n = threadIdx.x;
  dst[(size_t)w * DIM * DIM + (size_t)n * DIM + k] =
      (_Float16)srcs[w][(size_t)k * DIM + n];
}

// ---------------------------------------------------------------------------
// Kernel 1: GAT attention, one workgroup (8 waves) per batch element.
// ---------------------------------------------------------------------------
__global__ __launch_bounds__(256) void gat_kernel(
    const int* __restrict__ uids, const int* __restrict__ u_movies,
    const float* __restrict__ user_table, const float* __restrict__ item_table,
    const _Float16* __restrict__ wqT, const _Float16* __restrict__ wkT,
    const _Float16* __restrict__ wvT, const _Float16* __restrict__ woT,
    const float* __restrict__ bq, const float* __restrict__ bk,
    const float* __restrict__ bv, const float* __restrict__ bo,
    _Float16* __restrict__ aggh) {
  __shared__ _Float16 Xn[64 * DIM];   // neighbor rows (padded to 64)
  __shared__ _Float16 Xq[16 * DIM];   // node row, rows 1..15 zero
  __shared__ _Float16 Ks[64 * DIM];
  __shared__ _Float16 Vs[64 * DIM];
  __shared__ _Float16 Ys[16 * DIM];   // y row, rows 1..15 zero
  __shared__ float qf[DIM];
  __shared__ float sc[64];
  __shared__ float sred[2];

  const int tid  = threadIdx.x;
  const int lane = tid & 31;
  const int wave = tid >> 5;
  const int b    = blockIdx.x;
  const int uid  = uids[b];

  // Phase 0: gather node + neighbors into f16 LDS, zero padding rows.
  for (int idx = tid; idx < 16 * DIM; idx += 256) {
    int r = idx >> 7, dd = idx & 127;
    float v = (r == 0) ? user_table[(size_t)uid * DIM + dd] : 0.f;
    Xq[idx] = (_Float16)v;
    Ys[idx] = (_Float16)0.f;
  }
  for (int idx = tid; idx < 64 * 32; idx += 256) {
    int h = idx >> 5, c4 = idx & 31;
    float4 v = make_float4(0.f, 0.f, 0.f, 0.f);
    if (h < NH) {
      int mid = u_movies[((size_t)b * NH + h) * 2];
      v = *(const float4*)(item_table + (size_t)mid * DIM + c4 * 4);
    }
    _Float16* p = &Xn[h * DIM + c4 * 4];
    p[0] = (_Float16)v.x; p[1] = (_Float16)v.y;
    p[2] = (_Float16)v.z; p[3] = (_Float16)v.w;
  }
  __syncthreads();

  // Phase 1: q = relu(node @ Wq + bq); one 16-col tile per wave.
  {
    const int n0 = wave * 16;
    v8f c = {};
#pragma unroll
    for (int kc = 0; kc < 4; ++kc) {
      v16h a  = load_fragA(Xq, 0, kc * 32, lane);
      v16h bb = load_fragB(wqT, kc * 32, n0, lane);
      c = wmma16x16x32(a, bb, c);
    }
    if (lane < 16) {                       // row 0 lives in c[0] of lanes 0..15
      float v = c[0] + bq[n0 + lane];
      qf[n0 + lane] = v > 0.f ? v : 0.f;
    }
  }

  // Phase 2: K and V projections (A fragment shared between both WMMAs).
  for (int t = wave; t < 32; t += 8) {
    const int rt = t >> 3, nt = t & 7;
    const int row0 = rt * 16, n0 = nt * 16;
    v8f ck = {}, cv = {};
#pragma unroll
    for (int kc = 0; kc < 4; ++kc) {
      v16h a   = load_fragA(Xn, row0, kc * 32, lane);
      v16h bk_ = load_fragB(wkT, kc * 32, n0, lane);
      ck = wmma16x16x32(a, bk_, ck);
      v16h bv_ = load_fragB(wvT, kc * 32, n0, lane);
      cv = wmma16x16x32(a, bv_, cv);
    }
    store_tile_relu(Ks, row0, n0, bk, ck, lane);
    store_tile_relu(Vs, row0, n0, bv, cv, lane);
  }
  __syncthreads();

  // Phase 3: scores = q . k_h / sqrt(D)
  if (tid < 64) {
    float s = -1e30f;
    if (tid < NH) {
      const _Float16* kr = &Ks[tid * DIM];
      s = 0.f;
#pragma unroll 4
      for (int k = 0; k < DIM; ++k) s += qf[k] * (float)kr[k];
      s *= 0.08838834764831845f;  // 1/sqrt(128)
    }
    sc[tid] = s;
  }
  __syncthreads();

  // Phase 4: softmax over the 50 valid scores (tiny; single thread reduce).
  if (tid == 0) {
    float mx = -1e30f;
    for (int h = 0; h < NH; ++h) mx = fmaxf(mx, sc[h]);
    float sum = 0.f;
    for (int h = 0; h < NH; ++h) sum += __expf(sc[h] - mx);
    sred[0] = mx; sred[1] = 1.f / sum;
  }
  __syncthreads();
  if (tid < 64) sc[tid] = (tid < NH) ? __expf(sc[tid] - sred[0]) * sred[1] : 0.f;
  __syncthreads();

  // Phase 5: y = p @ V  (1 x 128)
  if (tid < DIM) {
    float acc = 0.f;
#pragma unroll 2
    for (int h = 0; h < NH; ++h) acc += sc[h] * (float)Vs[h * DIM + tid];
    Ys[tid] = (_Float16)acc;   // row 0
  }
  __syncthreads();

  // Phase 6: agg = relu(y @ Wo + bo) -> workspace (f16)
  {
    const int n0 = wave * 16;
    v8f c = {};
#pragma unroll
    for (int kc = 0; kc < 4; ++kc) {
      v16h a  = load_fragA(Ys, 0, kc * 32, lane);
      v16h bb = load_fragB(woT, kc * 32, n0, lane);
      c = wmma16x16x32(a, bb, c);
    }
    if (lane < 16) {
      float v = c[0] + bo[n0 + lane];
      aggh[(size_t)b * DIM + n0 + lane] = (_Float16)(v > 0.f ? v : 0.f);
    }
  }
}

// ---------------------------------------------------------------------------
// Kernel 2: SEM attention, one workgroup per 16 batch elements.
// keys per element = [node, agg] -> 32 rows = 2 WMMA row-tiles.
// ---------------------------------------------------------------------------
__global__ __launch_bounds__(256) void sem_kernel(
    const int* __restrict__ uids, const float* __restrict__ user_table,
    const _Float16* __restrict__ aggh,
    const _Float16* __restrict__ wqT, const _Float16* __restrict__ wkT,
    const _Float16* __restrict__ wvT, const _Float16* __restrict__ woT,
    const float* __restrict__ bq, const float* __restrict__ bk,
    const float* __restrict__ bv, const float* __restrict__ bo,
    float* __restrict__ out) {
  __shared__ _Float16 Xn[16 * DIM];    // node rows of the 16 elements
  __shared__ _Float16 Keys[32 * DIM];  // row 2i = node_i, 2i+1 = agg_i
  __shared__ _Float16 Ks[32 * DIM];
  __shared__ _Float16 Vs[32 * DIM];
  __shared__ _Float16 Ys[16 * DIM];
  __shared__ float qf[16 * DIM];
  __shared__ float sc[16][2];
  __shared__ float pr[16][2];

  const int tid  = threadIdx.x;
  const int lane = tid & 31;
  const int wave = tid >> 5;
  const int b0   = blockIdx.x * 16;

  for (int idx = tid; idx < 16 * DIM; idx += 256) {
    int i = idx >> 7, dd = idx & 127;
    int uid = uids[b0 + i];
    _Float16 nh = (_Float16)user_table[(size_t)uid * DIM + dd];
    Xn[idx] = nh;
    Keys[(2 * i) * DIM + dd]     = nh;
    Keys[(2 * i + 1) * DIM + dd] = aggh[(size_t)(b0 + i) * DIM + dd];
  }
  __syncthreads();

  // q projection (16 real rows), keep f32 for scores.
  {
    const int n0 = wave * 16;
    v8f c = {};
#pragma unroll
    for (int kc = 0; kc < 4; ++kc) {
      v16h a  = load_fragA(Xn, 0, kc * 32, lane);
      v16h bb = load_fragB(wqT, kc * 32, n0, lane);
      c = wmma16x16x32(a, bb, c);
    }
    int n = lane & 15, m0 = (lane & 16) ? 8 : 0;
    float bvq = bq[n0 + n];
#pragma unroll
    for (int r = 0; r < 8; ++r) {
      float v = c[r] + bvq;
      qf[(m0 + r) * DIM + n0 + n] = v > 0.f ? v : 0.f;
    }
  }

  // K and V projections: 2 row-tiles x 8 col-tiles.
  for (int t = wave; t < 16; t += 8) {
    const int rt = t >> 3, nt = t & 7;
    const int row0 = rt * 16, n0 = nt * 16;
    v8f ck = {}, cv = {};
#pragma unroll
    for (int kc = 0; kc < 4; ++kc) {
      v16h a   = load_fragA(Keys, row0, kc * 32, lane);
      v16h bk_ = load_fragB(wkT, kc * 32, n0, lane);
      ck = wmma16x16x32(a, bk_, ck);
      v16h bv_ = load_fragB(wvT, kc * 32, n0, lane);
      cv = wmma16x16x32(a, bv_, cv);
    }
    store_tile_relu(Ks, row0, n0, bk, ck, lane);
    store_tile_relu(Vs, row0, n0, bv, cv, lane);
  }
  __syncthreads();

  // Scores (2 keys each) + softmax.
  if (tid < 32) {
    int i = tid >> 1, j = tid & 1;
    const float* q = &qf[i * DIM];
    const _Float16* kr = &Ks[(2 * i + j) * DIM];
    float s = 0.f;
#pragma unroll 4
    for (int k = 0; k < DIM; ++k) s += q[k] * (float)kr[k];
    sc[i][j] = s * 0.08838834764831845f;
  }
  __syncthreads();
  if (tid < 16) {
    float s0 = sc[tid][0], s1 = sc[tid][1];
    float mx = fmaxf(s0, s1);
    float e0 = __expf(s0 - mx), e1 = __expf(s1 - mx);
    float inv = 1.f / (e0 + e1);
    pr[tid][0] = e0 * inv; pr[tid][1] = e1 * inv;
  }
  __syncthreads();

  // y = p0*v0 + p1*v1 per element.
  for (int idx = tid; idx < 16 * DIM; idx += 256) {
    int i = idx >> 7, dd = idx & 127;
    float y = pr[i][0] * (float)Vs[(2 * i) * DIM + dd] +
              pr[i][1] * (float)Vs[(2 * i + 1) * DIM + dd];
    Ys[idx] = (_Float16)y;
  }
  __syncthreads();

  // out = relu(y @ Wo + bo), fp32 to d_out.
  {
    const int n0 = wave * 16;
    v8f c = {};
#pragma unroll
    for (int kc = 0; kc < 4; ++kc) {
      v16h a  = load_fragA(Ys, 0, kc * 32, lane);
      v16h bb = load_fragB(woT, kc * 32, n0, lane);
      c = wmma16x16x32(a, bb, c);
    }
    int n = lane & 15, m0 = (lane & 16) ? 8 : 0;
    float bvo = bo[n0 + n];
#pragma unroll
    for (int r = 0; r < 8; ++r) {
      float v = c[r] + bvo;
      out[(size_t)(b0 + m0 + r) * DIM + n0 + n] = v > 0.f ? v : 0.f;
    }
  }
}

extern "C" void kernel_launch(void* const* d_in, const int* in_sizes, int n_in,
                              void* d_out, int out_size, void* d_ws, size_t ws_size,
                              hipStream_t stream) {
  (void)in_sizes; (void)n_in; (void)out_size; (void)ws_size;
  const int*   uids       = (const int*)d_in[0];
  const int*   u_movies   = (const int*)d_in[2];
  const float* user_table = (const float*)d_in[3];
  const float* item_table = (const float*)d_in[5];
  const float* gat_wq = (const float*)d_in[6];  const float* gat_bq = (const float*)d_in[7];
  const float* gat_wk = (const float*)d_in[8];  const float* gat_bk = (const float*)d_in[9];
  const float* gat_wv = (const float*)d_in[10]; const float* gat_bv = (const float*)d_in[11];
  const float* gat_wo = (const float*)d_in[12]; const float* gat_bo = (const float*)d_in[13];
  const float* sem_wq = (const float*)d_in[14]; const float* sem_bq = (const float*)d_in[15];
  const float* sem_wk = (const float*)d_in[16]; const float* sem_bk = (const float*)d_in[17];
  const float* sem_wv = (const float*)d_in[18]; const float* sem_bv = (const float*)d_in[19];
  const float* sem_wo = (const float*)d_in[20]; const float* sem_bo = (const float*)d_in[21];

  _Float16* wT   = (_Float16*)d_ws;            // 8 * 128*128 f16 transposed weights
  _Float16* aggh = wT + 8 * DIM * DIM;         // BATCH * 128 f16 agg rows

  prep_weights<<<8 * DIM, DIM, 0, stream>>>(gat_wq, gat_wk, gat_wv, gat_wo,
                                            sem_wq, sem_wk, sem_wv, sem_wo, wT);

  gat_kernel<<<BATCH, 256, 0, stream>>>(
      uids, u_movies, user_table, item_table,
      wT + 0 * DIM * DIM, wT + 1 * DIM * DIM, wT + 2 * DIM * DIM, wT + 3 * DIM * DIM,
      gat_bq, gat_bk, gat_bv, gat_bo, aggh);

  sem_kernel<<<BATCH / 16, 256, 0, stream>>>(
      uids, user_table, aggh,
      wT + 4 * DIM * DIM, wT + 5 * DIM * DIM, wT + 6 * DIM * DIM, wT + 7 * DIM * DIM,
      sem_bq, sem_bk, sem_bv, sem_bo, (float*)d_out);
}
